// GNN_Model_82532091560221
// MI455X (gfx1250) — compile-verified, matrix operand order checked by reading, not run on hardware
//
#include <hip/hip_runtime.h>

// ---------------------------------------------------------------------------
// Types / constants
// ---------------------------------------------------------------------------
typedef __attribute__((ext_vector_type(16))) _Float16 v16h;
typedef __attribute__((ext_vector_type(8)))  _Float16 v8h;
typedef __attribute__((ext_vector_type(8)))  float    v8f;

#define NPATHS  32768
#define NLINKS  32768
#define NQUEUES 98304
#define MAXLP   16
#define TITERS  4
#define RU      256
#define HSTRIDE 40   // padded row stride (halves) for h staging, conflict-free

// ---------------------------------------------------------------------------
// WMMA helpers (gfx1250 wave32, 16x16x32 f16 -> f32)
// ---------------------------------------------------------------------------
__device__ __forceinline__ v8f wmma16(v16h a, v16h b, v8f c) {
  return __builtin_amdgcn_wmma_f32_16x16x32_f16(false, a, false, b, (short)0, c,
                                                false, false);
}

// K index held by element i of an A/B operand for this lane (ISA 16-bit 16x32
// layout: VGPR0-3 = K 0..7 (+8 for upper half-wave), VGPR4-7 = K 16..23 (+8)).
__device__ __forceinline__ int abK(int lane, int i) {
  return ((i & 8) << 1) + ((lane & 16) >> 1) + (i & 7);
}

__device__ __forceinline__ v8f zero8() {
  v8f x;
#pragma unroll
  for (int i = 0; i < 8; ++i) x[i] = 0.f;
  return x;
}
__device__ __forceinline__ v16h zero16() {
  v16h x;
#pragma unroll
  for (int i = 0; i < 16; ++i) x[i] = (_Float16)0.f;
  return x;
}
__device__ __forceinline__ v16h cat8(v8h lo, v8h hi) {
  v16h a;
#pragma unroll
  for (int i = 0; i < 8; ++i) { a[i] = lo[i]; a[8 + i] = hi[i]; }
  return a;
}
__device__ __forceinline__ void put4(v16h& a, int o, float4 v) {
  a[o + 0] = (_Float16)v.x; a[o + 1] = (_Float16)v.y;
  a[o + 2] = (_Float16)v.z; a[o + 3] = (_Float16)v.w;
}

// Pre-packed B operand: packed[(tile*32 + lane)*16 + i]; one 32B read per lane
// (lowers to 2x ds_load_b128 from LDS).
__device__ __forceinline__ v16h loadBP(const _Float16* packed, int tile,
                                       int lane) {
  const v8h* p = (const v8h*)(packed + ((tile << 5) + lane) * 16);
  return cat8(p[0], p[1]);
}

// A operand from a row-major f32 row: two contiguous 8-float runs (32B aligned)
// -> 4x global_load_b128 + cvt.
__device__ __forceinline__ v16h loadA32v(const float* row, int lane) {
  int base = (lane & 16) >> 1;
  v16h a;
  put4(a, 0,  *(const float4*)(row + base));
  put4(a, 4,  *(const float4*)(row + base + 4));
  put4(a, 8,  *(const float4*)(row + base + 16));
  put4(a, 12, *(const float4*)(row + base + 20));
  return a;
}

// h C-layout (2 x v8f) -> f16 A-layout via per-wave LDS staging (row stride
// HSTRIDE halves). 16 b16 stores + 2 b128 loads per lane; wave-local DS is
// in-order, fences stop compiler reordering.
__device__ __forceinline__ v16h hC2A16(_Float16* lds, const v8f* h, int lane) {
  int c0 = lane & 15, rb = (lane & 16) >> 1;
#pragma unroll
  for (int j = 0; j < 2; ++j)
#pragma unroll
    for (int i = 0; i < 8; ++i)
      lds[(rb + i) * HSTRIDE + j * 16 + c0] = (_Float16)h[j][i];
  asm volatile("s_wait_dscnt 0" ::: "memory");
  int r = lane & 15;
  v8h lo = *(const v8h*)(lds + r * HSTRIDE + rb);
  v8h hi = *(const v8h*)(lds + r * HSTRIDE + 16 + rb);
  asm volatile("s_wait_dscnt 0" ::: "memory");
  return cat8(lo, hi);
}

// GRU elementwise combine in C layout; bs = {br0,br1,bz0,bz1,bi0,bi1,bh0,bh1}
// precomputed per lane. Branch-free masked update keeps EXEC uniform.
__device__ __forceinline__ void gruCombine(const v8f* acc, const v8f* hn,
                                           v8f* h, const float* bs, int mask) {
#pragma unroll
  for (int j = 0; j < 2; ++j) {
#pragma unroll
    for (int i = 0; i < 8; ++i) {
      float r  = 1.f / (1.f + __expf(-(acc[j][i] + bs[j])));
      float z  = 1.f / (1.f + __expf(-(acc[2 + j][i] + bs[2 + j])));
      float n  = tanhf(acc[4 + j][i] + bs[4 + j] + r * (hn[j][i] + bs[6 + j]));
      float hv = (1.f - z) * n + z * h[j][i];
      h[j][i]  = ((mask >> i) & 1) ? hv : h[j][i];
    }
  }
}

__device__ __forceinline__ void biasPrep(const float* __restrict__ bih,
                                         const float* __restrict__ bhh,
                                         int lane, float* bs) {
  int c0 = lane & 15;
#pragma unroll
  for (int j = 0; j < 2; ++j) {
    int c = j * 16 + c0;
    bs[j]     = bih[c] + bhh[c];
    bs[2 + j] = bih[32 + c] + bhh[32 + c];
    bs[4 + j] = bih[64 + c];
    bs[6 + j] = bhh[64 + c];
  }
}

// ---------------------------------------------------------------------------
// Setup: pack weights into WMMA lane layout (f16)
// ---------------------------------------------------------------------------
__global__ void k_prep_weights(const float* __restrict__ pWih,
                               const float* __restrict__ pWhh,
                               const float* __restrict__ lWih,
                               const float* __restrict__ lWhh,
                               const float* __restrict__ qWih,
                               const float* __restrict__ qWhh,
                               const float* __restrict__ W1,
                               const float* __restrict__ W2,
                               _Float16* pWihP, _Float16* pWhhP,
                               _Float16* lWihP, _Float16* lWhhP,
                               _Float16* qWihP, _Float16* qWhhP,
                               _Float16* w1p, _Float16* w2p) {
  int tid = blockIdx.x * blockDim.x + threadIdx.x;
  int stride = gridDim.x * blockDim.x;
  // path Wih: torch layout [96][64]; 2 K-chunks x 6 N-tiles
  for (int idx = tid; idx < 12 * 512; idx += stride) {
    int tile = idx >> 9, r = idx & 511, lane = r >> 4, i = r & 15;
    int kc = tile / 6, nt = tile % 6;
    int k = kc * 32 + abK(lane, i), n = nt * 16 + (lane & 15);
    pWihP[idx] = (_Float16)pWih[n * 64 + k];
  }
  // 32-din gate matrices [96][32]; 6 N-tiles each
  for (int idx = tid; idx < 6 * 512; idx += stride) {
    int tile = idx >> 9, r = idx & 511, lane = r >> 4, i = r & 15;
    int k = abK(lane, i), n = tile * 16 + (lane & 15);
    pWhhP[idx] = (_Float16)pWhh[n * 32 + k];
    lWihP[idx] = (_Float16)lWih[n * 32 + k];
    lWhhP[idx] = (_Float16)lWhh[n * 32 + k];
    qWihP[idx] = (_Float16)qWih[n * 32 + k];
    qWhhP[idx] = (_Float16)qWhh[n * 32 + k];
  }
  // W1: [32][256]; 16 N-tiles
  for (int idx = tid; idx < 16 * 512; idx += stride) {
    int tile = idx >> 9, r = idx & 511, lane = r >> 4, i = r & 15;
    int k = abK(lane, i), n = tile * 16 + (lane & 15);
    w1p[idx] = (_Float16)W1[k * RU + n];
  }
  // W2: [256][256]; 8 K-chunks x 16 N-tiles (tile = kc*16+nt)
  for (int idx = tid; idx < 128 * 512; idx += stride) {
    int tile = idx >> 9, r = idx & 511, lane = r >> 4, i = r & 15;
    int kc = tile >> 4, nt = tile & 15;
    int k = kc * 32 + abK(lane, i), n = nt * 16 + (lane & 15);
    w2p[idx] = (_Float16)W2[k * RU + n];
  }
}

__global__ void k_init_states(const float* __restrict__ traffic,
                              const float* __restrict__ packets,
                              const float* __restrict__ capacity,
                              const float* __restrict__ weight,
                              const int* __restrict__ policy,
                              const int* __restrict__ priority,
                              float* pstate, float* lstate, float* qstate) {
  int tid = blockIdx.x * blockDim.x + threadIdx.x;
  int stride = gridDim.x * blockDim.x;
  for (int i = tid; i < NPATHS * 32; i += stride) {
    int r = i >> 5, c = i & 31;
    pstate[i] = (c == 0) ? traffic[r] : (c == 1 ? packets[r] : 0.f);
  }
  for (int i = tid; i < NLINKS * 32; i += stride) {
    int r = i >> 5, c = i & 31;
    float v = 0.f;
    if (c == 0) v = capacity[r];
    else if (c <= 3) v = (policy[r] == (c - 1)) ? 1.f : 0.f;
    lstate[i] = v;
  }
  for (int i = tid; i < NQUEUES * 32; i += stride) {
    int r = i >> 5, c = i & 31;
    float v = 0.f;
    if (c < 3) v = (priority[r] == c) ? 1.f : 0.f;
    else if (c == 3) v = weight[r];
    qstate[i] = v;
  }
}

__global__ void k_zero_i32(int* p, int n) {
  int i = blockIdx.x * blockDim.x + threadIdx.x;
  if (i < n) p[i] = 0;
}
__global__ void k_zero_f32(float* p, int n) {
  int i = blockIdx.x * blockDim.x + threadIdx.x;
  if (i < n) p[i] = 0.f;
}

__global__ void k_build_edges(const int* __restrict__ path_ids,
                              const int* __restrict__ l_p_s,
                              int* edge_table, int* lens, int E) {
  int e = blockIdx.x * blockDim.x + threadIdx.x;
  if (e >= E) return;
  int p = path_ids[e], t = l_p_s[e];
  edge_table[p * MAXLP + t] = e;
  atomicMax(&lens[p], t + 1);
}

__global__ void k_scatter_sum(const float* __restrict__ pstate,
                              const int* __restrict__ pid,
                              const int* __restrict__ q2p,
                              float* psum, int total) {
  int idx = blockIdx.x * blockDim.x + threadIdx.x;
  if (idx >= total) return;
  int e = idx >> 5, c = idx & 31;
  atomicAdd(&psum[q2p[e] * 32 + c], pstate[pid[e] * 32 + c]);
}

// ---------------------------------------------------------------------------
// Path GRU: 16 masked steps over hop sequences. One wave per 16-path tile.
// ---------------------------------------------------------------------------
__global__ void __launch_bounds__(256)
k_path_gru(const float* __restrict__ qstate, const float* __restrict__ lstate,
           float* __restrict__ pstate, const int* __restrict__ edge_table,
           const int* __restrict__ lens, const int* __restrict__ q2p,
           const int* __restrict__ l2p, const _Float16* __restrict__ wih,
           const _Float16* __restrict__ whh, const float* __restrict__ bih,
           const float* __restrict__ bhh) {
  __shared__ uint4 sWihR[768];                 // 12 packed tiles
  __shared__ uint4 sWhhR[384];                 // 6 packed tiles
  __shared__ uint4 sHR[8 * (16 * HSTRIDE * 2 / 16)];
  _Float16* sWih = (_Float16*)sWihR;
  _Float16* sWhh = (_Float16*)sWhhR;
  {
    uint4* d = sWihR; const uint4* s = (const uint4*)wih;
    for (int i = threadIdx.x; i < 768; i += 256) d[i] = s[i];
    d = sWhhR; s = (const uint4*)whh;
    for (int i = threadIdx.x; i < 384; i += 256) d[i] = s[i];
  }
  __syncthreads();

  int wave = threadIdx.x >> 5, lane = threadIdx.x & 31;
  int p0 = (blockIdx.x * 8 + wave) * 16;
  int c0 = lane & 15, rb = (lane & 16) >> 1;
  _Float16* myLds = (_Float16*)sHR + wave * 16 * HSTRIDE;

  float bs[8];
  biasPrep(bih, bhh, lane, bs);

  v8f h[2];
#pragma unroll
  for (int j = 0; j < 2; ++j)
#pragma unroll
    for (int i = 0; i < 8; ++i)
      h[j][i] = pstate[(p0 + rb + i) * 32 + j * 16 + c0];

  int mylen[8];
#pragma unroll
  for (int i = 0; i < 8; ++i) mylen[i] = lens[p0 + rb + i];
  int rowp = p0 + c0;            // row this lane gathers for A operands
  int glen = lens[rowp];

  for (int t = 0; t < MAXLP; ++t) {
    v16h hA = hC2A16(myLds, h, lane);

    v16h xa0 = zero16(), xa1 = zero16();
    if (t < glen) {
      int e = edge_table[rowp * MAXLP + t];
      const float* qrow = qstate + q2p[e] * 32;
      const float* lrow = lstate + l2p[e] * 32;
      put4(xa0, 0,  *(const float4*)(qrow + rb));
      put4(xa0, 4,  *(const float4*)(qrow + rb + 4));
      put4(xa0, 8,  *(const float4*)(qrow + rb + 16));
      put4(xa0, 12, *(const float4*)(qrow + rb + 20));
      put4(xa1, 0,  *(const float4*)(lrow + rb));
      put4(xa1, 4,  *(const float4*)(lrow + rb + 4));
      put4(xa1, 8,  *(const float4*)(lrow + rb + 16));
      put4(xa1, 12, *(const float4*)(lrow + rb + 20));
    }

    v8f acc[6], hn[2];
#pragma unroll
    for (int nt = 0; nt < 6; ++nt) {
      v8f a = wmma16(xa0, loadBP(sWih, nt, lane), zero8());
      acc[nt] = wmma16(xa1, loadBP(sWih, 6 + nt, lane), a);
    }
#pragma unroll
    for (int nt = 0; nt < 4; ++nt)
      acc[nt] = wmma16(hA, loadBP(sWhh, nt, lane), acc[nt]);
#pragma unroll
    for (int j = 0; j < 2; ++j)
      hn[j] = wmma16(hA, loadBP(sWhh, 4 + j, lane), zero8());

    int mask = 0;
#pragma unroll
    for (int i = 0; i < 8; ++i) mask |= (t < mylen[i]) ? (1 << i) : 0;
    gruCombine(acc, hn, h, bs, mask);
  }

#pragma unroll
  for (int j = 0; j < 2; ++j)
#pragma unroll
    for (int i = 0; i < 8; ++i)
      pstate[(p0 + rb + i) * 32 + j * 16 + c0] = h[j][i];
}

// ---------------------------------------------------------------------------
// Queue update: single GRU cell on (path_sum, queue_state).
// ---------------------------------------------------------------------------
__global__ void __launch_bounds__(256)
k_queue_cell(const float* __restrict__ psum, float* __restrict__ qstate,
             const _Float16* __restrict__ wih, const _Float16* __restrict__ whh,
             const float* __restrict__ bih, const float* __restrict__ bhh) {
  __shared__ uint4 sWihR[384];
  __shared__ uint4 sWhhR[384];
  _Float16* sWih = (_Float16*)sWihR;
  _Float16* sWhh = (_Float16*)sWhhR;
  {
    const uint4* s0 = (const uint4*)wih;
    const uint4* s1 = (const uint4*)whh;
    for (int i = threadIdx.x; i < 384; i += 256) {
      sWihR[i] = s0[i];
      sWhhR[i] = s1[i];
    }
  }
  __syncthreads();

  int wave = threadIdx.x >> 5, lane = threadIdx.x & 31;
  int q0 = (blockIdx.x * 8 + wave) * 16;
  int c0 = lane & 15, rb = (lane & 16) >> 1;

  float bs[8];
  biasPrep(bih, bhh, lane, bs);

  v8f h[2];
#pragma unroll
  for (int j = 0; j < 2; ++j)
#pragma unroll
    for (int i = 0; i < 8; ++i)
      h[j][i] = qstate[(q0 + rb + i) * 32 + j * 16 + c0];

  v16h xa = loadA32v(psum + (q0 + c0) * 32, lane);
  v16h hA = loadA32v(qstate + (q0 + c0) * 32, lane);

  v8f acc[6], hn[2];
#pragma unroll
  for (int nt = 0; nt < 6; ++nt)
    acc[nt] = wmma16(xa, loadBP(sWih, nt, lane), zero8());
#pragma unroll
  for (int nt = 0; nt < 4; ++nt)
    acc[nt] = wmma16(hA, loadBP(sWhh, nt, lane), acc[nt]);
#pragma unroll
  for (int j = 0; j < 2; ++j)
    hn[j] = wmma16(hA, loadBP(sWhh, 4 + j, lane), zero8());

  gruCombine(acc, hn, h, bs, 0xFF);

#pragma unroll
  for (int j = 0; j < 2; ++j)
#pragma unroll
    for (int i = 0; i < 8; ++i)
      qstate[(q0 + rb + i) * 32 + j * 16 + c0] = h[j][i];
}

// ---------------------------------------------------------------------------
// Link update: 3-step GRU over each link's queues; h0 = path_state
// (replicates the reference's quirk). lens_l == 3 always in practice.
// ---------------------------------------------------------------------------
__global__ void __launch_bounds__(256)
k_link_gru(const float* __restrict__ qstate, const float* __restrict__ pstate,
           float* __restrict__ lstate, const _Float16* __restrict__ wih,
           const _Float16* __restrict__ whh, const float* __restrict__ bih,
           const float* __restrict__ bhh) {
  __shared__ uint4 sWihR[384];
  __shared__ uint4 sWhhR[384];
  __shared__ uint4 sHR[8 * (16 * HSTRIDE * 2 / 16)];
  _Float16* sWih = (_Float16*)sWihR;
  _Float16* sWhh = (_Float16*)sWhhR;
  {
    const uint4* s0 = (const uint4*)wih;
    const uint4* s1 = (const uint4*)whh;
    for (int i = threadIdx.x; i < 384; i += 256) {
      sWihR[i] = s0[i];
      sWhhR[i] = s1[i];
    }
  }
  __syncthreads();

  int wave = threadIdx.x >> 5, lane = threadIdx.x & 31;
  int l0 = (blockIdx.x * 8 + wave) * 16;
  int c0 = lane & 15, rb = (lane & 16) >> 1;
  _Float16* myLds = (_Float16*)sHR + wave * 16 * HSTRIDE;

  float bs[8];
  biasPrep(bih, bhh, lane, bs);

  v8f h[2];
#pragma unroll
  for (int j = 0; j < 2; ++j)
#pragma unroll
    for (int i = 0; i < 8; ++i)
      h[j][i] = pstate[(l0 + rb + i) * 32 + j * 16 + c0];

  for (int t = 0; t < 3; ++t) {
    v16h hA = hC2A16(myLds, h, lane);
    v16h xa = loadA32v(qstate + ((l0 + c0) * 3 + t) * 32, lane);

    v8f acc[6], hn[2];
#pragma unroll
    for (int nt = 0; nt < 6; ++nt)
      acc[nt] = wmma16(xa, loadBP(sWih, nt, lane), zero8());
#pragma unroll
    for (int nt = 0; nt < 4; ++nt)
      acc[nt] = wmma16(hA, loadBP(sWhh, nt, lane), acc[nt]);
#pragma unroll
    for (int j = 0; j < 2; ++j)
      hn[j] = wmma16(hA, loadBP(sWhh, 4 + j, lane), zero8());

    gruCombine(acc, hn, h, bs, 0xFF);
  }

#pragma unroll
  for (int j = 0; j < 2; ++j)
#pragma unroll
    for (int i = 0; i < 8; ++i)
      lstate[(l0 + rb + i) * 32 + j * 16 + c0] = h[j][i];
}

// ---------------------------------------------------------------------------
// Readout
// ---------------------------------------------------------------------------
__global__ void __launch_bounds__(256)
k_readout1(const float* __restrict__ pstate, const _Float16* __restrict__ w1,
           const float* __restrict__ b1, _Float16* __restrict__ h1) {
  __shared__ uint4 sWR[1024];                  // 16 packed tiles
  _Float16* sW = (_Float16*)sWR;
  {
    const uint4* s = (const uint4*)w1;
    for (int i = threadIdx.x; i < 1024; i += 256) sWR[i] = s[i];
  }
  __syncthreads();

  int wave = threadIdx.x >> 5, lane = threadIdx.x & 31;
  int m0 = (blockIdx.x * 8 + wave) * 16;
  int c0 = lane & 15, rb = (lane & 16) >> 1;

  v16h a = loadA32v(pstate + (m0 + c0) * 32, lane);
#pragma unroll
  for (int nt = 0; nt < 16; ++nt) {
    v8f acc = wmma16(a, loadBP(sW, nt, lane), zero8());
    int col = nt * 16 + c0;
    float b = b1[col];
#pragma unroll
    for (int i = 0; i < 8; ++i) {
      float v = acc[i] + b;
      h1[(m0 + rb + i) * RU + col] = (_Float16)(v > 0.f ? v : 0.f);
    }
  }
}

__global__ void __launch_bounds__(256)
k_readout2(const _Float16* __restrict__ h1, const _Float16* __restrict__ w2,
           const float* __restrict__ b2, _Float16* __restrict__ h2) {
  __shared__ uint4 sWR[1024];                  // one 16-tile K-chunk at a time
  _Float16* sW = (_Float16*)sWR;
  int wave = threadIdx.x >> 5, lane = threadIdx.x & 31;
  int m0 = (blockIdx.x * 8 + wave) * 16;
  int c0 = lane & 15, rb = (lane & 16) >> 1;

  v8f acc[16];
#pragma unroll
  for (int nt = 0; nt < 16; ++nt) acc[nt] = zero8();

  for (int kc = 0; kc < 8; ++kc) {
    __syncthreads();
    {
      const uint4* s = (const uint4*)w2 + kc * 1024;
      for (int i = threadIdx.x; i < 1024; i += 256) sWR[i] = s[i];
    }
    __syncthreads();

    const _Float16* xrow = h1 + (m0 + c0) * RU + kc * 32;
    v16h a = cat8(*(const v8h*)(xrow + rb), *(const v8h*)(xrow + 16 + rb));
#pragma unroll
    for (int nt = 0; nt < 16; ++nt)
      acc[nt] = wmma16(a, loadBP(sW, nt, lane), acc[nt]);
  }

#pragma unroll
  for (int nt = 0; nt < 16; ++nt) {
    int col = nt * 16 + c0;
    float b = b2[col];
#pragma unroll
    for (int i = 0; i < 8; ++i) {
      float v = acc[nt][i] + b;
      h2[(m0 + rb + i) * RU + col] = (_Float16)(v > 0.f ? v : 0.f);
    }
  }
}

__global__ void k_readout3(const _Float16* __restrict__ h2,
                           const float* __restrict__ w3,
                           const float* __restrict__ b3,
                           float* __restrict__ out) {
  int p = blockIdx.x * blockDim.x + threadIdx.x;
  if (p >= NPATHS) return;
  float acc = b3[0];
  const v8h* row = (const v8h*)(h2 + p * RU);
#pragma unroll 4
  for (int c = 0; c < RU / 8; ++c) {
    v8h v = row[c];
#pragma unroll
    for (int k = 0; k < 8; ++k) acc += (float)v[k] * w3[c * 8 + k];
  }
  out[p] = acc;
}

// ---------------------------------------------------------------------------
// Host launcher
// ---------------------------------------------------------------------------
extern "C" void kernel_launch(void* const* d_in, const int* in_sizes, int n_in,
                              void* d_out, int out_size, void* d_ws,
                              size_t ws_size, hipStream_t stream) {
  const float* traffic  = (const float*)d_in[0];
  const float* packets  = (const float*)d_in[1];
  const float* capacity = (const float*)d_in[2];
  const float* weight   = (const float*)d_in[3];
  const int*   policy   = (const int*)d_in[4];
  const int*   priority = (const int*)d_in[5];
  const int*   path_ids = (const int*)d_in[6];
  const int*   l_p_s    = (const int*)d_in[8];
  const int*   l2p      = (const int*)d_in[10];
  const int*   q2p      = (const int*)d_in[11];
  const int E = in_sizes[6];

  const float* pWih = (const float*)d_in[19];
  const float* pWhh = (const float*)d_in[20];
  const float* pbih = (const float*)d_in[21];
  const float* pbhh = (const float*)d_in[22];
  const float* lWih = (const float*)d_in[23];
  const float* lWhh = (const float*)d_in[24];
  const float* lbih = (const float*)d_in[25];
  const float* lbhh = (const float*)d_in[26];
  const float* qWih = (const float*)d_in[27];
  const float* qWhh = (const float*)d_in[28];
  const float* qbih = (const float*)d_in[29];
  const float* qbhh = (const float*)d_in[30];
  const float* W1   = (const float*)d_in[31];
  const float* b1   = (const float*)d_in[32];
  const float* W2   = (const float*)d_in[33];
  const float* b2   = (const float*)d_in[34];
  const float* W3   = (const float*)d_in[35];
  const float* b3   = (const float*)d_in[36];

  char* w = (char*)d_ws;
  auto carve = [&](size_t bytes) {
    void* p = (void*)w;
    w += (bytes + 255) & ~(size_t)255;
    return p;
  };
  float* pstate = (float*)carve((size_t)NPATHS * 32 * 4);
  float* lstate = (float*)carve((size_t)NLINKS * 32 * 4);
  float* qstate = (float*)carve((size_t)NQUEUES * 32 * 4);
  float* psum   = (float*)carve((size_t)NQUEUES * 32 * 4);
  int* edge_table = (int*)carve((size_t)NPATHS * MAXLP * 4);
  int* lens       = (int*)carve((size_t)NPATHS * 4);
  _Float16* pWihP = (_Float16*)carve(12 * 512 * 2);
  _Float16* pWhhP = (_Float16*)carve(6 * 512 * 2);
  _Float16* lWihP = (_Float16*)carve(6 * 512 * 2);
  _Float16* lWhhP = (_Float16*)carve(6 * 512 * 2);
  _Float16* qWihP = (_Float16*)carve(6 * 512 * 2);
  _Float16* qWhhP = (_Float16*)carve(6 * 512 * 2);
  _Float16* w1p   = (_Float16*)carve(16 * 512 * 2);
  _Float16* w2p   = (_Float16*)carve(128 * 512 * 2);
  _Float16* h1    = (_Float16*)carve((size_t)NPATHS * RU * 2);
  _Float16* h2    = (_Float16*)carve((size_t)NPATHS * RU * 2);

  // --- setup ---
  k_prep_weights<<<64, 256, 0, stream>>>(pWih, pWhh, lWih, lWhh, qWih, qWhh,
                                         W1, W2, pWihP, pWhhP, lWihP, lWhhP,
                                         qWihP, qWhhP, w1p, w2p);
  k_init_states<<<1024, 256, 0, stream>>>(traffic, packets, capacity, weight,
                                          policy, priority, pstate, lstate,
                                          qstate);
  k_zero_i32<<<(NPATHS + 255) / 256, 256, 0, stream>>>(lens, NPATHS);
  k_build_edges<<<(E + 255) / 256, 256, 0, stream>>>(path_ids, l_p_s,
                                                     edge_table, lens, E);

  // --- message passing ---
  for (int it = 0; it < TITERS; ++it) {
    k_path_gru<<<NPATHS / 128, 256, 0, stream>>>(
        qstate, lstate, pstate, edge_table, lens, q2p, l2p, pWihP, pWhhP,
        pbih, pbhh);
    k_zero_f32<<<(NQUEUES * 32 + 255) / 256, 256, 0, stream>>>(psum,
                                                               NQUEUES * 32);
    k_scatter_sum<<<(E * 32 + 255) / 256, 256, 0, stream>>>(pstate, path_ids,
                                                            q2p, psum, E * 32);
    k_queue_cell<<<NQUEUES / 128, 256, 0, stream>>>(psum, qstate, qWihP, qWhhP,
                                                    qbih, qbhh);
    k_link_gru<<<NLINKS / 128, 256, 0, stream>>>(qstate, pstate, lstate, lWihP,
                                                 lWhhP, lbih, lbhh);
  }

  // --- readout ---
  k_readout1<<<NPATHS / 128, 256, 0, stream>>>(pstate, w1p, b1, h1);
  k_readout2<<<NPATHS / 128, 256, 0, stream>>>(h1, w2p, b2, h2);
  k_readout3<<<NPATHS / 256, 256, 0, stream>>>(h2, W3, b3, (float*)d_out);
}
